// MotionModule_8340826489077
// MI455X (gfx1250) — matrix operand order repaired and checked by reading, not compiled, not run on hardware
//
#include <hip/hip_runtime.h>
#include <hip/hip_bf16.h>
#include <math.h>

// MotionModule (temporal transformer) for MI455X / gfx1250.
// Large GEMMs: v_wmma_f32_16x16x32_f16 with double-buffered LDS tiles filled by
// GLOBAL_LOAD_ASYNC_TO_LDS_B128 (ASYNCcnt-tracked DMA, no VGPR round trip).
// Weights pre-transposed to [N][K]; both WMMA fragments load as two
// ds_load_b128 per lane from 80B-stride (bank-conflict-free) LDS tiles.

typedef _Float16 v16h __attribute__((ext_vector_type(16)));
typedef _Float16 v8h  __attribute__((ext_vector_type(8)));
typedef float    v8f  __attribute__((ext_vector_type(8)));

#define TOKENS 32768   // B*F*H*W = 2*16*1024
#define CDIM   320
#define FRAMES 16
#define NHEADS 8
#define DHEAD  40
#define INNER  1280
#define FFOUT  2560

// Low 32 bits of a generic pointer to __shared__ == LDS byte offset
// (flat LDS aperture truncates to addr[31:0], ISA 10.2).
__device__ __forceinline__ unsigned lds_off(const void* p) {
  return (unsigned)(size_t)p;
}

// ---------------- fp32 -> f16 convert ----------------
__global__ void k_cvt(const float* __restrict__ in, _Float16* __restrict__ out, int n) {
  int i = blockIdx.x * 256 + threadIdx.x;
  if (i < n) out[i] = (_Float16)in[i];
}

// ---------------- fp32 [K][N] -> f16 transposed [N][K] ----------------
__global__ void k_cvt_t(const float* __restrict__ src, _Float16* __restrict__ dst,
                        int K, int N) {
  int idx = blockIdx.x * 256 + threadIdx.x;
  if (idx >= K * N) return;
  int n = idx / K, k = idx % K;                 // write-contiguous
  dst[idx] = (_Float16)src[(size_t)k * N + n];
}

// ---------------- GroupNorm(32, eps 1e-6) -> f16 tokens [bf, hw, c] ----------------
__global__ __launch_bounds__(256) void k_gn(const float* __restrict__ x,
                                            const float* __restrict__ gw,
                                            const float* __restrict__ gb,
                                            _Float16* __restrict__ out) {
  int bf = blockIdx.x, g = blockIdx.y;              // 32 images x 32 groups
  const float* base = x + ((size_t)bf * CDIM + (size_t)g * 10) * 1024;
  __shared__ float s1[256], s2[256];
  float s = 0.f, ss = 0.f;
  for (int i = threadIdx.x; i < 10240; i += 256) { float v = base[i]; s += v; ss += v * v; }
  s1[threadIdx.x] = s; s2[threadIdx.x] = ss; __syncthreads();
  for (int off = 128; off > 0; off >>= 1) {
    if ((int)threadIdx.x < off) { s1[threadIdx.x] += s1[threadIdx.x + off];
                                  s2[threadIdx.x] += s2[threadIdx.x + off]; }
    __syncthreads();
  }
  float mean = s1[0] * (1.f / 10240.f);
  float var  = s2[0] * (1.f / 10240.f) - mean * mean;
  float inv  = rsqrtf(var + 1e-6f);
  for (int i = threadIdx.x; i < 10240; i += 256) {
    int cc = i >> 10, p = i & 1023, ch = g * 10 + cc;
    float v = (base[i] - mean) * inv * gw[ch] + gb[ch];
    out[((size_t)bf * 1024 + p) * CDIM + ch] = (_Float16)v;
  }
}

// ---------------- LayerNorm (eps 1e-5) [+ temporal PE] -> f16 ----------------
__global__ __launch_bounds__(128) void k_ln(const float* __restrict__ in,
                                            const float* __restrict__ w,
                                            const float* __restrict__ b,
                                            _Float16* __restrict__ out, int addPE) {
  int t = blockIdx.x;                                 // token id
  const float* row = in + (size_t)t * CDIM;
  __shared__ float s1[128], s2[128];
  float vals[3]; float s = 0.f, ss = 0.f;
  for (int i = 0; i < 3; i++) {
    int c = threadIdx.x + i * 128;
    float v = (c < CDIM) ? row[c] : 0.f;
    vals[i] = v; s += v; ss += v * v;
  }
  s1[threadIdx.x] = s; s2[threadIdx.x] = ss; __syncthreads();
  for (int off = 64; off > 0; off >>= 1) {
    if ((int)threadIdx.x < off) { s1[threadIdx.x] += s1[threadIdx.x + off];
                                  s2[threadIdx.x] += s2[threadIdx.x + off]; }
    __syncthreads();
  }
  float mean = s1[0] * (1.f / CDIM);
  float var  = s2[0] * (1.f / CDIM) - mean * mean;
  float inv  = rsqrtf(var + 1e-5f);
  int f = (t >> 10) & 15;                             // frame index of this token
  for (int i = 0; i < 3; i++) {
    int c = threadIdx.x + i * 128;
    if (c < CDIM) {
      float v = (vals[i] - mean) * inv * w[c] + b[c];
      if (addPE) {
        float div = __expf(-(float)(c & ~1) * (logf(10000.f) / (float)CDIM));
        float ang = (float)f * div;
        v += (c & 1) ? __cosf(ang) : __sinf(ang);
      }
      out[(size_t)t * CDIM + c] = (_Float16)v;
    }
  }
}

// ---------------- WMMA GEMM: C[M,N] = A[M,K](f16) @ Bt[N,K](f16) (+bias)(+res) ----------------
// 128 threads = 4 waves. Block tile 64x64, K-step 32, double-buffered LDS filled
// with global_load_async_to_lds_b128 (ASYNCcnt). Each wave: one A frag x four
// B frags -> 4 WMMAs per K-step; all fragments are two ds_load_b128 per lane.
__global__ __launch_bounds__(128) void k_gemm_wmma(const _Float16* __restrict__ A,
                                                   const _Float16* __restrict__ Bt,
                                                   const float* __restrict__ bias,
                                                   const float* res, float* Cout,
                                                   int M, int N, int K) {
  __shared__ __align__(16) _Float16 sA[2][64][40];   // 10 KB: row m, K-contig
  __shared__ __align__(16) _Float16 sB[2][64][40];   // 10 KB: col n, K-contig
  const int tid  = threadIdx.x;
  const int wave = tid >> 5;
  const int lane = tid & 31;
  const int bm = blockIdx.y * 64;
  const int bn = blockIdx.x * 64;

  // cooperative tile mapping: thread -> 32 contiguous bytes of one row/col
  const int r  = tid >> 1;
  const int ch = (tid & 1) * 16;
  const _Float16* gA = A  + (size_t)(bm + r) * K + ch;
  const _Float16* gB = Bt + (size_t)(bn + r) * K + ch;

  v8f acc[4];
  for (int nt = 0; nt < 4; ++nt)
    for (int j = 0; j < 8; ++j) acc[nt][j] = 0.f;

  auto issue = [&](int buf, int k0) {
    unsigned la = lds_off(&sA[buf][r][ch]);
    unsigned lb = lds_off(&sB[buf][r][ch]);
    const _Float16* pa = gA + k0;
    const _Float16* pb = gB + k0;
    asm volatile("global_load_async_to_lds_b128 %0, %1, off"
                 :: "v"(la), "v"(pa) : "memory");
    asm volatile("global_load_async_to_lds_b128 %0, %1, off offset:16"
                 :: "v"(la), "v"(pa) : "memory");   // offset adds to LDS and mem addr
    asm volatile("global_load_async_to_lds_b128 %0, %1, off"
                 :: "v"(lb), "v"(pb) : "memory");
    asm volatile("global_load_async_to_lds_b128 %0, %1, off offset:16"
                 :: "v"(lb), "v"(pb) : "memory");
  };

  int buf = 0;
  issue(0, 0);                                   // prime buffer 0
  for (int k0 = 0; k0 < K; k0 += 32) {
    int kn = (k0 + 32 < K) ? (k0 + 32) : 0;      // always prefetch (tail harmless)
    issue(buf ^ 1, kn);
    // async loads complete in order per wave: <=4 outstanding => current tile done
    asm volatile("s_wait_asynccnt 0x4" ::: "memory");
    __syncthreads();

    // A fragment (16x32 f16): lanes 0-15 row m K{0-7,16-23}; lanes 16-31 K{8-15,24-31}
    v16h af;
    {
      int row   = wave * 16 + (lane & 15);
      int khalf = lane >> 4;
      v8h g0 = *(const v8h*)&sA[buf][row][khalf * 8];
      v8h g1 = *(const v8h*)&sA[buf][row][16 + khalf * 8];
      af = __builtin_shufflevector(g0, g1, 0,1,2,3,4,5,6,7,8,9,10,11,12,13,14,15);
    }
    // four B fragments (32x16): lane = column; lanes 0-15 K{0-7,16-23}; lanes 16-31 K{8-15,24-31}
    for (int nt = 0; nt < 4; ++nt) {
      int col = nt * 16 + (lane & 15);
      int kb  = (lane >> 4) * 8;
      v8h b0 = *(const v8h*)&sB[buf][col][kb];
      v8h b1 = *(const v8h*)&sB[buf][col][kb + 16];
      v16h bfrag = __builtin_shufflevector(b0, b1, 0,1,2,3,4,5,6,7,8,9,10,11,12,13,14,15);
      acc[nt] = __builtin_amdgcn_wmma_f32_16x16x32_f16(
          false, af, false, bfrag, (short)0, acc[nt], false, false);
    }
    __syncthreads();                             // all waves done reading buf
    buf ^= 1;
  }

  // Epilogue: C/D layout -> m = wave*16 + v + 8*(lane>>4), n = nt*16 + (lane&15)
  int mbase = bm + wave * 16 + 8 * (lane >> 4);
  int nbase = bn + (lane & 15);
  for (int nt = 0; nt < 4; ++nt) {
    int n = nbase + nt * 16;
    float bv = bias ? bias[n] : 0.f;
    for (int v = 0; v < 8; ++v) {
      size_t idx = (size_t)(mbase + v) * N + n;
      float val = acc[nt][v] + bv;
      if (res) val += res[idx];
      Cout[idx] = val;
    }
  }
}

// ---------------- temporal attention core: seq=16, 8 heads x dh=40 ----------------
// One block per (b,d) site; threads = (head, query). K/V rows staged in LDS.
__global__ __launch_bounds__(128) void k_attn(const float* __restrict__ Q,
                                              const float* __restrict__ Kb,
                                              const float* __restrict__ V,
                                              _Float16* __restrict__ O) {
  int n = blockIdx.x;                 // 0..2047
  int b = n >> 10, d = n & 1023;
  __shared__ float Ks[FRAMES][CDIM];  // 20 KB
  __shared__ float Vs[FRAMES][CDIM];  // 20 KB
  for (int idx = threadIdx.x; idx < FRAMES * CDIM; idx += 128) {
    int f = idx / CDIM, c = idx % CDIM;
    size_t tok = ((size_t)(b * FRAMES + f) * 1024 + d);
    Ks[f][c] = Kb[tok * CDIM + c];
    Vs[f][c] = V[tok * CDIM + c];
  }
  __syncthreads();
  int h = threadIdx.x >> 4, q = threadIdx.x & 15;
  size_t tq = ((size_t)(b * FRAMES + q) * 1024 + d);
  float qv[DHEAD];
  for (int j = 0; j < DHEAD; ++j) qv[j] = Q[tq * CDIM + h * DHEAD + j];
  float sc[FRAMES]; float mx = -1e30f;
  const float scale = 0.15811388300841898f;  // 40^-0.5
  for (int f = 0; f < FRAMES; ++f) {
    float s = 0.f;
    for (int j = 0; j < DHEAD; ++j) s += qv[j] * Ks[f][h * DHEAD + j];
    s *= scale; sc[f] = s; mx = fmaxf(mx, s);
  }
  float denom = 0.f;
  for (int f = 0; f < FRAMES; ++f) { sc[f] = __expf(sc[f] - mx); denom += sc[f]; }
  float rden = 1.f / denom;
  for (int j = 0; j < DHEAD; ++j) {
    float o = 0.f;
    for (int f = 0; f < FRAMES; ++f) o += sc[f] * Vs[f][h * DHEAD + j];
    O[tq * CDIM + h * DHEAD + j] = (_Float16)(o * rden);
  }
}

// ---------------- GEGLU: a * gelu_exact(gate) -> f16 ----------------
__global__ void k_geglu(const float* __restrict__ y, _Float16* __restrict__ out) {
  size_t idx = (size_t)blockIdx.x * 256 + threadIdx.x;
  if (idx >= (size_t)TOKENS * INNER) return;
  size_t m = idx / INNER; int j = (int)(idx % INNER);
  float a = y[m * FFOUT + j];
  float g = y[m * FFOUT + INNER + j];
  float gl = 0.5f * g * (1.f + erff(g * 0.70710678118654752f));
  out[idx] = (_Float16)(a * gl);
}

// ---------------- final: out[bf,c,p] = tok[bf,p,c] + x[bf,c,p] ----------------
__global__ void k_final(const float* __restrict__ t, const float* __restrict__ x,
                        float* __restrict__ out) {
  size_t idx = (size_t)blockIdx.x * 256 + threadIdx.x;
  if (idx >= (size_t)32 * CDIM * 1024) return;
  size_t bf = idx / (CDIM * 1024);
  int r = (int)(idx % (CDIM * 1024));
  int c = r >> 10, p = r & 1023;
  out[idx] = x[idx] + t[(bf * 1024 + p) * CDIM + c];
}

extern "C" void kernel_launch(void* const* d_in, const int* in_sizes, int n_in,
                              void* d_out, int out_size, void* d_ws, size_t ws_size,
                              hipStream_t stream) {
  (void)in_sizes; (void)n_in; (void)out_size; (void)ws_size;
  const float* x     = (const float*)d_in[0];
  const float* gn_w  = (const float*)d_in[1];
  const float* gn_b  = (const float*)d_in[2];
  const float* pin_w = (const float*)d_in[3];
  const float* pin_b = (const float*)d_in[4];
  const float* ln_w[2]  = {(const float*)d_in[5],  (const float*)d_in[12]};
  const float* ln_b[2]  = {(const float*)d_in[6],  (const float*)d_in[13]};
  const float* wq[2]    = {(const float*)d_in[7],  (const float*)d_in[14]};
  const float* wk[2]    = {(const float*)d_in[8],  (const float*)d_in[15]};
  const float* wv[2]    = {(const float*)d_in[9],  (const float*)d_in[16]};
  const float* wo[2]    = {(const float*)d_in[10], (const float*)d_in[17]};
  const float* bo[2]    = {(const float*)d_in[11], (const float*)d_in[18]};
  const float* ffln_w = (const float*)d_in[19];
  const float* ffln_b = (const float*)d_in[20];
  const float* ff_w1  = (const float*)d_in[21];
  const float* ff_b1  = (const float*)d_in[22];
  const float* ff_w2  = (const float*)d_in[23];
  const float* ff_b2  = (const float*)d_in[24];
  const float* pout_w = (const float*)d_in[25];
  const float* pout_b = (const float*)d_in[26];

  size_t off = 0;
  auto alloc = [&](size_t bytes) -> void* {
    off = (off + 255) & ~(size_t)255;
    void* p = (char*)d_ws + off; off += bytes; return p;
  };
  const size_t SQ = (size_t)CDIM * CDIM;             // 102400
  _Float16* pin16  = (_Float16*)alloc(SQ * 2);
  _Float16* wq16[2], *wk16[2], *wv16[2], *wo16[2];
  for (int i = 0; i < 2; ++i) {
    wq16[i] = (_Float16*)alloc(SQ * 2); wk16[i] = (_Float16*)alloc(SQ * 2);
    wv16[i] = (_Float16*)alloc(SQ * 2); wo16[i] = (_Float16*)alloc(SQ * 2);
  }
  _Float16* ffw1_16 = (_Float16*)alloc((size_t)CDIM * FFOUT * 2);
  _Float16* ffw2_16 = (_Float16*)alloc((size_t)INNER * CDIM * 2);
  _Float16* pout16  = (_Float16*)alloc(SQ * 2);
  float*    TOK = (float*)alloc((size_t)TOKENS * CDIM * 4);     // residual stream
  _Float16* A16 = (_Float16*)alloc((size_t)TOKENS * INNER * 2); // GEMM-A staging
  float*    QB  = (float*)alloc((size_t)TOKENS * CDIM * 4);
  float*    KB  = (float*)alloc((size_t)TOKENS * CDIM * 4);
  float*    VB  = (float*)alloc((size_t)TOKENS * CDIM * 4);
  float*    Y   = (float*)alloc((size_t)TOKENS * FFOUT * 4);    // FF1 out / pout tmp

  auto cvtT = [&](const float* s, _Float16* d, int K, int N) {   // [K][N] -> f16 [N][K]
    int n = K * N;
    k_cvt_t<<<(unsigned)((n + 255) / 256), 256, 0, stream>>>(s, d, K, N);
  };
  auto gemm = [&](const _Float16* A, const _Float16* W, const float* bias,
                  const float* res, float* out, int M, int N, int K) {
    dim3 grid(N / 64, M / 64);
    k_gemm_wmma<<<grid, 128, 0, stream>>>(A, W, bias, res, out, M, N, K);
  };

  // weights -> f16, transposed to [N][K]
  cvtT(pin_w, pin16, CDIM, CDIM);
  for (int i = 0; i < 2; ++i) {
    cvtT(wq[i], wq16[i], CDIM, CDIM); cvtT(wk[i], wk16[i], CDIM, CDIM);
    cvtT(wv[i], wv16[i], CDIM, CDIM); cvtT(wo[i], wo16[i], CDIM, CDIM);
  }
  cvtT(ff_w1, ffw1_16, CDIM, FFOUT);
  cvtT(ff_w2, ffw2_16, INNER, CDIM);
  cvtT(pout_w, pout16, CDIM, CDIM);

  // GroupNorm -> tokens(f16), proj_in -> TOK
  k_gn<<<dim3(32, 32), 256, 0, stream>>>(x, gn_w, gn_b, A16);
  gemm(A16, pin16, pin_b, nullptr, TOK, TOKENS, CDIM, CDIM);

  // two temporal attention blocks
  for (int i = 0; i < 2; ++i) {
    k_ln<<<TOKENS, 128, 0, stream>>>(TOK, ln_w[i], ln_b[i], A16, 1 /*+PE*/);
    gemm(A16, wq16[i], nullptr, nullptr, QB, TOKENS, CDIM, CDIM);
    gemm(A16, wk16[i], nullptr, nullptr, KB, TOKENS, CDIM, CDIM);
    gemm(A16, wv16[i], nullptr, nullptr, VB, TOKENS, CDIM, CDIM);
    k_attn<<<2048, 128, 0, stream>>>(QB, KB, VB, A16);
    gemm(A16, wo16[i], bo[i], TOK, TOK, TOKENS, CDIM, CDIM);  // out-proj + residual
  }

  // GEGLU feed-forward
  k_ln<<<TOKENS, 128, 0, stream>>>(TOK, ffln_w, ffln_b, A16, 0);
  gemm(A16, ffw1_16, ff_b1, nullptr, Y, TOKENS, FFOUT, CDIM);
  k_geglu<<<(unsigned)(((size_t)TOKENS * INNER + 255) / 256), 256, 0, stream>>>(Y, A16);
  gemm(A16, ffw2_16, ff_b2, TOK, TOK, TOKENS, CDIM, INNER);   // + residual

  // proj_out + spatial residual
  k_cvt<<<(unsigned)(((size_t)TOKENS * CDIM + 255) / 256), 256, 0, stream>>>(
      TOK, A16, TOKENS * CDIM);
  gemm(A16, pout16, pout_b, nullptr, Y, TOKENS, CDIM, CDIM);
  k_final<<<(unsigned)(((size_t)32 * CDIM * 1024 + 255) / 256), 256, 0, stream>>>(
      Y, x, (float*)d_out);
}
